// DependencySelfAttention_80711025426870
// MI455X (gfx1250) — matrix operand order repaired and checked by reading, not compiled, not run on hardware
//
#include <hip/hip_runtime.h>

typedef __attribute__((ext_vector_type(16))) _Float16 v16h;
typedef __attribute__((ext_vector_type(8)))  _Float16 v8h;
typedef __attribute__((ext_vector_type(4)))  _Float16 v4h;
typedef __attribute__((ext_vector_type(8)))  float    v8f;

#define B_  16
#define S_  2048
#define D_  256
#define BM  32
#define BN  32
#define VT_PAD 48   // halfs per V^T row (96B: 16B aligned, bank-skewed)
#define AT_PAD 48   // halfs per attn row

union U16h { v16h v; v8h p[2]; _Float16 h[16]; };
union U4h  { v4h  v; _Float16 h[4]; };

// ---- 8-element loaders -> v8h (f16 path: raw b128; f32 path: load + cvt) ----
__device__ __forceinline__ v8h load8h(const _Float16* p) {
    return *(const v8h*)__builtin_assume_aligned(p, 16);
}
__device__ __forceinline__ v8h load8h(const float* p) {
    float4 a = *(const float4*)p;
    float4 b = *(const float4*)(p + 4);
    union { v8h v; _Float16 h[8]; } u;
    u.h[0]=(_Float16)a.x; u.h[1]=(_Float16)a.y; u.h[2]=(_Float16)a.z; u.h[3]=(_Float16)a.w;
    u.h[4]=(_Float16)b.x; u.h[5]=(_Float16)b.y; u.h[6]=(_Float16)b.z; u.h[7]=(_Float16)b.w;
    return u.v;
}

// 16-consecutive-element fragment (B layout: lanes>=16 already offset by caller)
template <typename T>
__device__ __forceinline__ void load_kfrag(const T* row, int kbase, U16h& u) {
    u.p[0] = load8h(row + kbase);
    u.p[1] = load8h(row + kbase + 8);
}
__device__ __forceinline__ void load_ldsfrag(const _Float16* base, U16h& u) {
    u.p[0] = *(const v8h*)__builtin_assume_aligned(base, 16);
    u.p[1] = *(const v8h*)__builtin_assume_aligned(base + 8, 16);
}

// ---- prep: P f32 -> f16 row-major copy in workspace ----
__global__ __launch_bounds__(256)
void cvt_f32_f16(const float* __restrict__ src, _Float16* __restrict__ dst, int n) {
    int i = (blockIdx.x * blockDim.x + threadIdx.x) * 8;
    if (i < n) {
        v8h v = load8h(src + i);
        *(v8h*)__builtin_assume_aligned(dst + i, 16) = v;
    }
}

template <typename T>
__global__ __launch_bounds__(128)
void dep_attn_fused(const T*   __restrict__ P,
                    const int* __restrict__ pm,
                    const int* __restrict__ dep,
                    float*     __restrict__ out)
{
    __shared__ __align__(16) _Float16 sVT[D_ * VT_PAD];   // V^T: [d][j] f16
    __shared__ __align__(16) _Float16 sAT[BM * AT_PAD];   // attn*dep tile f16, row-major
    __shared__ float sPMax[2][BM];
    __shared__ float sPSum[2][BM];
    __shared__ float sRowM[BM];
    __shared__ float sRowL[BM];
    __shared__ float sRowScale[BM];

    const int b    = blockIdx.y;
    const int i0   = blockIdx.x * BM;
    const int tid  = threadIdx.x;
    const int lane = tid & 31;
    const int wave = tid >> 5;
    const int rw   = wave >> 1;   // row half of the 32-row tile
    const int cw   = wave & 1;    // column half (S-tile) == D half (O accum)
    const int ln   = lane & 15;
    const int hi   = lane >> 4;   // 0: lanes 0-15, 1: lanes 16-31

    const T* Pb = P + (size_t)b * S_ * D_;

    // ---- preload Q A-fragments: rows i0 + rw*16 + ln, K = 0..255 (8 chunks) ----
    v16h qa[8];
    {
        const T* qrow = Pb + (size_t)(i0 + rw * 16 + ln) * D_;
        #pragma unroll
        for (int kc = 0; kc < 8; ++kc) {
            const int kb = kc * 32 + hi * 16;      // A layout: lanes>=16 hold K+16
            U16h u;
            u.p[0] = load8h(qrow + kb);
            u.p[1] = load8h(qrow + kb + 8);
            qa[kc] = u.v;
        }
    }

    const v8f vzero = {};
    v8f o[8];
    #pragma unroll
    for (int dt = 0; dt < 8; ++dt) o[dt] = vzero;

    if (tid < BM) { sRowM[tid] = -3.0e38f; sRowL[tid] = 0.0f; }

    // staging geometry: each thread stages a 4-row x 16-d patch of V^T
    const int jq = (tid & 7) * 4;     // j quad base
    const int dc = (tid >> 3) * 16;   // d chunk base

    for (int jt = 0; jt < S_ / BN; ++jt) {
        const int j0 = jt * BN;
        __syncthreads();   // previous iteration's LDS reads complete

        // ---- prefetch next tile's dep rows (streamed once) ----
        if (ln == 0 && jt + 1 < S_ / BN) {
            #pragma unroll
            for (int v = 0; v < 8; ++v) {
                const int irow = rw * 16 + v + hi * 8;
                __builtin_prefetch(&dep[((size_t)(b * S_ + i0 + irow)) * S_ + j0 + BN + cw * 16], 0, 0);
            }
        }

        // ---- stage V tile into LDS transposed as f16 (b64 packed stores) ----
        {
            U16h rowv[4];
            #pragma unroll
            for (int r4 = 0; r4 < 4; ++r4) {
                const T* vr = Pb + (size_t)(j0 + jq + r4) * D_ + dc;
                rowv[r4].p[0] = load8h(vr);
                rowv[r4].p[1] = load8h(vr + 8);
            }
            #pragma unroll
            for (int e = 0; e < 16; ++e) {
                U4h q;
                q.h[0] = rowv[0].h[e];
                q.h[1] = rowv[1].h[e];
                q.h[2] = rowv[2].h[e];
                q.h[3] = rowv[3].h[e];
                *(v4h*)__builtin_assume_aligned(&sVT[(dc + e) * VT_PAD + jq], 8) = q.v;
            }
        }

        // ---- S tile: 16x16 = Q(rw) . K(cw)^T over K=256 (2-deep pipelined) ----
        v8f s = vzero;
        {
            const T* krow = Pb + (size_t)(j0 + cw * 16 + ln) * D_;
            U16h uk[2];
            load_kfrag(krow, hi * 16, uk[0]);      // kc = 0
            #pragma unroll
            for (int kc = 0; kc < 8; ++kc) {
                if (kc + 1 < 8)
                    load_kfrag(krow, (kc + 1) * 32 + hi * 16, uk[(kc + 1) & 1]);
                s = __builtin_amdgcn_wmma_f32_16x16x32_f16(
                        false, qa[kc], false, uk[kc & 1].v, (short)0, s, false, false);
            }
        }

        // ---- column mask & masked partial row max ----
        const int   jcol = j0 + cw * 16 + ln;
        const float mj   = (float)pm[b * S_ + jcol];

        #pragma unroll
        for (int v = 0; v < 8; ++v) {
            float x = (mj != 0.0f) ? s[v] : -3.0e38f;
            x = fmaxf(x, __shfl_xor(x, 1, 32));
            x = fmaxf(x, __shfl_xor(x, 2, 32));
            x = fmaxf(x, __shfl_xor(x, 4, 32));
            x = fmaxf(x, __shfl_xor(x, 8, 32));
            if (ln == 0) sPMax[cw][rw * 16 + v + hi * 8] = x;
        }
        __syncthreads();

        // ---- wave0: running max & rescale factor ----
        if (tid < BM) {
            const int r = tid;
            float pmx  = fmaxf(sPMax[0][r], sPMax[1][r]);
            float mOld = sRowM[r];
            float mNew = fmaxf(fmaxf(mOld, pmx), -1.0e30f);
            sRowScale[r] = __expf(mOld - mNew);
            sRowM[r]     = mNew;
        }
        __syncthreads();

        // ---- w = m_j * exp(s - M); partial row sums; attn*dep -> LDS f16 ----
        #pragma unroll
        for (int v = 0; v < 8; ++v) {
            const int irow = rw * 16 + v + hi * 8;
            const float M  = sRowM[irow];
            float w = (mj != 0.0f) ? __expf(s[v] - M) : 0.0f;
            float t = w;
            t += __shfl_xor(t, 1, 32);
            t += __shfl_xor(t, 2, 32);
            t += __shfl_xor(t, 4, 32);
            t += __shfl_xor(t, 8, 32);
            if (ln == 0) sPSum[cw][irow] = t;
            const int dv = __builtin_nontemporal_load(
                               &dep[((size_t)(b * S_ + i0 + irow)) * S_ + jcol]);
            sAT[irow * AT_PAD + cw * 16 + ln] = (_Float16)(w * (float)dv);
        }
        __syncthreads();

        // ---- wave0: running sum update ----
        if (tid < BM) {
            const int r = tid;
            sRowL[r] = sRowL[r] * sRowScale[r] + sPSum[0][r] + sPSum[1][r];
        }

        // ---- per-row rescale factors (1.0 when max unchanged) ----
        float sc[8];
        #pragma unroll
        for (int v = 0; v < 8; ++v) sc[v] = sRowScale[rw * 16 + v + hi * 8];

        // ---- A-fragment: attn rows rw*16+ln over 32 tile columns ----
        U16h ua;
        {
            const _Float16* ar = &sAT[(rw * 16 + ln) * AT_PAD];
            ua.p[0] = *(const v8h*)__builtin_assume_aligned(ar + hi * 16, 16);
            ua.p[1] = *(const v8h*)__builtin_assume_aligned(ar + hi * 16 + 8, 16);
        }

        // ---- PV: 2-deep pipelined B-frags, rescale interleaved with WMMA ----
        {
            const _Float16* vbase = &sVT[(cw * 128 + ln) * VT_PAD + hi * 16];
            U16h ub[2];
            load_ldsfrag(vbase, ub[0]);            // dt = 0
            #pragma unroll
            for (int dt = 0; dt < 8; ++dt) {
                if (dt + 1 < 8)
                    load_ldsfrag(vbase + (dt + 1) * 16 * VT_PAD, ub[(dt + 1) & 1]);
                #pragma unroll
                for (int v = 0; v < 8; ++v) o[dt][v] *= sc[v];
                o[dt] = __builtin_amdgcn_wmma_f32_16x16x32_f16(
                            false, ua.v, false, ub[dt & 1].v, (short)0, o[dt], false, false);
            }
        }
    }

    __syncthreads();

    // ---- epilogue: O / (l + 1e-13), * row mask, NT store f32 ----
    float inv[8], rmk[8];
    #pragma unroll
    for (int v = 0; v < 8; ++v) {
        const int irow = rw * 16 + v + hi * 8;
        inv[v] = 1.0f / (sRowL[irow] + 1e-13f);
        rmk[v] = (float)pm[b * S_ + i0 + irow];
    }
    #pragma unroll
    for (int dt = 0; dt < 8; ++dt) {
        const int dcol = cw * 128 + dt * 16 + ln;
        #pragma unroll
        for (int v = 0; v < 8; ++v) {
            const int irow = rw * 16 + v + hi * 8;
            __builtin_nontemporal_store(o[dt][v] * inv[v] * rmk[v],
                                        &out[((size_t)b * S_ + i0 + irow) * D_ + dcol]);
        }
    }
}

extern "C" void kernel_launch(void* const* d_in, const int* in_sizes, int n_in,
                              void* d_out, int out_size, void* d_ws, size_t ws_size,
                              hipStream_t stream) {
    (void)in_sizes; (void)n_in; (void)out_size;
    const float* P   = (const float*)d_in[0];
    const int*   pm  = (const int*)d_in[1];
    const int*   dep = (const int*)d_in[2];
    float*       out = (float*)d_out;

    dim3 grid(S_ / BM, B_, 1);
    dim3 block(128, 1, 1);

    const int    n    = B_ * S_ * D_;
    const size_t need = (size_t)n * 2;   // f16 copy of P

    if (ws_size >= need) {
        _Float16* Ph = (_Float16*)d_ws;
        cvt_f32_f16<<<dim3(n / 8 / 256, 1, 1), dim3(256, 1, 1), 0, stream>>>(P, Ph, n);
        dep_attn_fused<_Float16><<<grid, block, 0, stream>>>(Ph, pm, dep, out);
    } else {
        dep_attn_fused<float><<<grid, block, 0, stream>>>(P, pm, dep, out);
    }
}
